// MWT_CZ_40415642255919
// MI455X (gfx1250) — compile-verified, matrix operand order checked by reading, not run on hardware
//
#include <hip/hip_runtime.h>
#include <math.h>

typedef __attribute__((ext_vector_type(16))) _Float16      v16h;
typedef __attribute__((ext_vector_type(8)))  float         v8f;
typedef __attribute__((ext_vector_type(8)))  unsigned int  v8u;

#define NB     32      // batch
#define NN     8192    // sequence length
#define CKD    64      // c*k channels
#define NLV    13      // log2(NN)
#define KSPLIT 4       // forward-DFT K-split factor

// ---------- helpers ----------
static __device__ __forceinline__ unsigned short f2h(float x){
  _Float16 h = (_Float16)x;
  return __builtin_bit_cast(unsigned short, h);
}
static __device__ __forceinline__ unsigned pk2h(float a, float b){
  auto h = __builtin_amdgcn_cvt_pkrtz(a, b);     // v_cvt_pk_f16_f32 -> __fp16x2
  return __builtin_bit_cast(unsigned, h);
}
static __device__ __forceinline__ v8f wmma_f16(v16h a, v16h b, v8f c){
  // v_wmma_f32_16x16x32_f16 : D = A(16x32) * B(32x16) + C (f32 accum)
  return __builtin_amdgcn_wmma_f32_16x16x32_f16(false, a, false, b, (short)0, c, false, false);
}

// Unguarded 32x16 f16 B-tile load (K contiguous in memory at tab+rowoff).
static __device__ __forceinline__ v16h load_tab_h(const unsigned short* __restrict__ tab,
                                                  long rowoff, int k0, int koff){
  v8u u;
#pragma unroll
  for (int h = 0; h < 2; ++h)
#pragma unroll
    for (int v = 0; v < 4; ++v){
      int k = k0 + h*16 + koff + 2*v;
      u[h*4+v] = *(const unsigned*)(tab + rowoff + k);
    }
  return __builtin_bit_cast(v16h, u);
}
// Guarded variant (tail / tiny-n levels only).
static __device__ __forceinline__ v16h load_tab_h_g(const unsigned short* __restrict__ tab,
                                                    long rowoff, int n, int k0, int koff){
  v8u u;
#pragma unroll
  for (int h = 0; h < 2; ++h)
#pragma unroll
    for (int v = 0; v < 4; ++v){
      int k = k0 + h*16 + koff + 2*v;
      unsigned w = 0u;
      if (k + 1 < n)      w = *(const unsigned*)(tab + rowoff + k);
      else if (k < n)     w = (unsigned)tab[rowoff + k];
      u[h*4+v] = w;
    }
  return __builtin_bit_cast(v16h, u);
}

// ---------- 1. per-level DFT basis tables (f16) ----------
// forward : ftc[m][t]= cos(2*pi*m*t/n), fts[m][t]= -sin(...)     (X = sum x*(cos - i sin))
// inverse : itc[t][m]= w_m*cos(...),    its[t][m]= -w_m*sin(...)  (x = sum Re*wc + Im*(-ws))
// rows with m >= lmodes are zero => truncated-mode semantics fall out of the GEMM.
__global__ void tables_kernel(unsigned short* __restrict__ ftc, unsigned short* __restrict__ fts,
                              unsigned short* __restrict__ itc, unsigned short* __restrict__ its,
                              int n, int lmodes){
  int total = 64 * n;
  for (int idx = blockIdx.x*blockDim.x + threadIdx.x; idx < total;
       idx += gridDim.x*blockDim.x){
    int m = idx / n, t = idx - m*n;
    float ang = 6.28318530717958647f * (float)((m*t) % n) / (float)n;
    float cc = cosf(ang), ss = sinf(ang);
    bool keep = (m < lmodes);
    ftc[m*n + t] = keep ? f2h(cc)  : (unsigned short)0;
    fts[m*n + t] = keep ? f2h(-ss) : (unsigned short)0;
    float w = ((m == 0) || (2*m == n)) ? (1.0f/(float)n) : (2.0f/(float)n);
    itc[t*64 + m] = keep ? f2h(w*cc)  : (unsigned short)0;
    its[t*64 + m] = keep ? f2h(-w*ss) : (unsigned short)0;
  }
}

// ---------- 2. wavelet decomposition: xa=[x_even,x_odd], d=xa@ec_d, s=xa@ec_s ----------
__global__ void decompose_kernel(const float* __restrict__ x,
                                 const float* __restrict__ ec_d,
                                 const float* __restrict__ ec_s,
                                 float* __restrict__ d, float* __restrict__ s, int n){
  __shared__ float ed[128], es[128];
  for (int i = threadIdx.x; i < 128; i += blockDim.x){ ed[i]=ec_d[i]; es[i]=ec_s[i]; }
  __syncthreads();
  long total = (long)NB * n * 8;                 // (b, t, c)
  for (long idx = (long)blockIdx.x*blockDim.x + threadIdx.x; idx < total;
       idx += (long)gridDim.x*blockDim.x){
    int  c  = (int)(idx & 7);
    long bt = idx >> 3;
    int  t  = (int)(bt % n);
    int  b  = (int)(bt / n);
    const float* x0 = x + (((long)b*(2*n) + 2*t)*CKD + c*8);
    float a[16];
#pragma unroll
    for (int j = 0; j < 8; ++j){ a[j] = x0[j]; a[8+j] = x0[CKD + j]; }
    float dv[8] = {0,0,0,0,0,0,0,0}, sv[8] = {0,0,0,0,0,0,0,0};
#pragma unroll
    for (int j = 0; j < 16; ++j)
#pragma unroll
      for (int k = 0; k < 8; ++k){ dv[k] += a[j]*ed[j*8+k]; sv[k] += a[j]*es[j*8+k]; }
    long off = ((long)b*n + t)*CKD + c*8;
#pragma unroll
    for (int k = 0; k < 8; ++k){ d[off+k] = dv[k]; s[off+k] = sv[k]; }
  }
}

// ---------- 3. forward truncated DFT (WMMA, K-split) ----------
// partial spectrum [z][b][m][i] = sum_{t in z-chunk} src[b][t][i]*tab[m][t]
__global__ void __launch_bounds__(256)
fwd_dft_kernel(const float* __restrict__ dsrc, const float* __restrict__ ssrc,
               const unsigned short* __restrict__ ftc, const unsigned short* __restrict__ fts,
               float* __restrict__ DFre, float* __restrict__ DFim,
               float* __restrict__ SFre, float* __restrict__ SFim, int n){
  int b = blockIdx.x;
  const float* src = blockIdx.y ? ssrc : dsrc;
  float* outRe = blockIdx.y ? SFre : DFre;
  float* outIm = blockIdx.y ? SFim : DFim;
  int z = blockIdx.z;

  int chunk = ((n + KSPLIT*32 - 1) / (KSPLIT*32)) * 32;   // 32-multiple chunk
  int tbeg  = z * chunk;
  int tend  = (tbeg + chunk < n) ? (tbeg + chunk) : n;
  if (tend < tbeg) tend = tbeg;                           // empty chunk -> zeros
  int tfull = tbeg + ((tend - tbeg) & ~31);               // == tend unless n < 32

  int wave = threadIdx.x >> 5, lane = threadIdx.x & 31;
  int it   = wave >> 1;                 // i-tile 0..3
  int mt0  = (wave & 1) * 2;            // m-tiles {mt0, mt0+1}
  int row  = it*16 + (lane & 15);       // A row: channel i
  int koff = (lane >> 4) * 8;           // 16-bit WMMA K sub-pattern
  long col0 = (long)(mt0*16 + (lane & 15)) * n;   // B row offset for m-tile 0
  long col1 = col0 + 16L*n;                       // m-tile 1

  v8f acc0re = {}, acc0im = {}, acc1re = {}, acc1im = {};
  const float* srcb = src + (long)b*n*CKD;

  for (int t0 = tbeg; t0 < tfull; t0 += 32){      // hot loop: no guards at all
    v8u au;
#pragma unroll
    for (int h = 0; h < 2; ++h)
#pragma unroll
      for (int v = 0; v < 4; ++v){
        int t = t0 + h*16 + koff + 2*v;
        au[h*4+v] = pk2h(srcb[(long)t*CKD + row], srcb[(long)(t+1)*CKD + row]);
      }
    v16h a = __builtin_bit_cast(v16h, au);
    v16h bc0 = load_tab_h(ftc, col0, t0, koff);
    v16h bs0 = load_tab_h(fts, col0, t0, koff);
    v16h bc1 = load_tab_h(ftc, col1, t0, koff);
    v16h bs1 = load_tab_h(fts, col1, t0, koff);
    acc0re = wmma_f16(a, bc0, acc0re);
    acc0im = wmma_f16(a, bs0, acc0im);            // fts already holds -sin
    acc1re = wmma_f16(a, bc1, acc1re);
    acc1im = wmma_f16(a, bs1, acc1im);
  }
  if (tfull < tend){                              // only for n < 32 levels
    int t0 = tfull;
    v8u au;
#pragma unroll
    for (int h = 0; h < 2; ++h)
#pragma unroll
      for (int v = 0; v < 4; ++v){
        int t = t0 + h*16 + koff + 2*v;
        float f0 = (t     < tend) ? srcb[(long)t*CKD + row]     : 0.0f;
        float f1 = (t + 1 < tend) ? srcb[(long)(t+1)*CKD + row] : 0.0f;
        au[h*4+v] = pk2h(f0, f1);
      }
    v16h a = __builtin_bit_cast(v16h, au);
    v16h bc0 = load_tab_h_g(ftc, col0, tend, t0, koff);
    v16h bs0 = load_tab_h_g(fts, col0, tend, t0, koff);
    v16h bc1 = load_tab_h_g(ftc, col1, tend, t0, koff);
    v16h bs1 = load_tab_h_g(fts, col1, tend, t0, koff);
    acc0re = wmma_f16(a, bc0, acc0re);
    acc0im = wmma_f16(a, bs0, acc0im);
    acc1re = wmma_f16(a, bc1, acc1re);
    acc1im = wmma_f16(a, bs1, acc1im);
  }
  // C tile (M=i, N=m) -> store [z][b][m][i]: 8 consecutive floats, 32B aligned
  int nlo = lane & 15, hi = lane >> 4;
  long ibase = (long)it*16 + hi*8;
  long mA = (long)mt0*16 + nlo, mB = mA + 16;
  long zb = (long)z*NB + b;
  *(v8f*)(outRe + (zb*64 + mA)*64 + ibase) = acc0re;
  *(v8f*)(outIm + (zb*64 + mA)*64 + ibase) = acc0im;
  *(v8f*)(outRe + (zb*64 + mB)*64 + ibase) = acc1re;
  *(v8f*)(outIm + (zb*64 + mB)*64 + ibase) = acc1im;
}

// ---------- 4. spectral mixing (exact f32): of[b,o,m] = sum_i xf[b,i,m]*W[i,o,m] ----------
// Also reduces the KSPLIT forward partials while staging into LDS.
__global__ void __launch_bounds__(64)
mix_kernel(const float* __restrict__ DFre, const float* __restrict__ DFim,
           const float* __restrict__ SFre, const float* __restrict__ SFim,
           const float* __restrict__ wAre, const float* __restrict__ wAim,
           const float* __restrict__ wBre, const float* __restrict__ wBim,
           const float* __restrict__ wCre, const float* __restrict__ wCim,
           float* __restrict__ UdFre, float* __restrict__ UdFim,
           float* __restrict__ UsFre, float* __restrict__ UsFim){
  int b = blockIdx.x, m = blockIdx.y, o = threadIdx.x;
  __shared__ float dre[64], dim_[64], sre[64], sim_[64];
  float adr = 0.f, adi = 0.f, asr = 0.f, asi = 0.f;
#pragma unroll
  for (int z = 0; z < KSPLIT; ++z){
    long base = (((long)z*NB + b)*64 + m)*64 + o;   // partials [z][b][m][i]
    adr += DFre[base]; adi += DFim[base];
    asr += SFre[base]; asi += SFim[base];
  }
  dre[o] = adr; dim_[o] = adi; sre[o] = asr; sim_[o] = asi;
  __syncthreads();
  float udre = 0.f, udim = 0.f, usre = 0.f, usim = 0.f;
  for (int i = 0; i < 64; ++i){
    long w = ((long)i*64 + o)*64 + m;              // W layout [i][o][m]
    float ar = wAre[w], ai = wAim[w];
    float br = wBre[w], bi = wBim[w];
    float cr = wCre[w], ci = wCim[w];
    float xr = dre[i], xi = dim_[i], yr = sre[i], yi = sim_[i];
    udre += xr*ar - xi*ai + yr*br - yi*bi;
    udim += xr*ai + xi*ar + yr*bi + yi*br;
    usre += xr*cr - xi*ci;
    usim += xr*ci + xi*cr;
  }
  long ob = ((long)b*64 + o)*64 + m;               // mixed layout [b][o][m]
  UdFre[ob] = udre; UdFim[ob] = udim;
  UsFre[ob] = usre; UsFim[ob] = usim;
}

// ---------- 5. inverse truncated DFT (WMMA): out[b][t][i] = sum_m Re*itc[t][m] + Im*its[t][m] ----------
__global__ void __launch_bounds__(256)
inv_dft_kernel(const float* __restrict__ UdFre, const float* __restrict__ UdFim,
               const float* __restrict__ UsFre, const float* __restrict__ UsFim,
               float* __restrict__ outUd, float* __restrict__ outUs,
               const unsigned short* __restrict__ itc, const unsigned short* __restrict__ its,
               int n){
  int b = blockIdx.x;
  const float* Re = blockIdx.y ? UsFre : UdFre;
  const float* Im = blockIdx.y ? UsFim : UdFim;
  float* out = blockIdx.y ? outUs : outUd;

  int wave = threadIdx.x >> 5, lane = threadIdx.x & 31;
  int it   = wave >> 1;                            // i-tile 0..3
  int t0   = blockIdx.z*32 + (wave & 1)*16;        // t-tile base
  int row  = it*16 + (lane & 15);                  // A row: channel o
  int colt = t0 + (lane & 15);                     // B column: time t
  int koff = (lane >> 4) * 8;

  const float* ReB = Re + (long)b*4096 + (long)row*64;
  const float* ImB = Im + (long)b*4096 + (long)row*64;
  v8f acc = {};
#pragma unroll
  for (int ks = 0; ks < 2; ++ks){                  // K = 64 modes, two 32-steps
    int m0 = ks*32;
    v8u aru, aiu;
#pragma unroll
    for (int h = 0; h < 2; ++h)
#pragma unroll
      for (int v = 0; v < 4; ++v){
        int m = m0 + h*16 + koff + 2*v;
        float2 rr = *(const float2*)(ReB + m);     // 8B-aligned (m even)
        float2 ii = *(const float2*)(ImB + m);
        aru[h*4+v] = pk2h(rr.x, rr.y);
        aiu[h*4+v] = pk2h(ii.x, ii.y);
      }
    v16h ar = __builtin_bit_cast(v16h, aru);
    v16h ai = __builtin_bit_cast(v16h, aiu);
    v8u bcu = {}, bsu = {};
    if (colt < n){
      long rb = (long)colt*64;
#pragma unroll
      for (int h = 0; h < 2; ++h)
#pragma unroll
        for (int v = 0; v < 4; ++v){
          int m = m0 + h*16 + koff + 2*v;
          bcu[h*4+v] = *(const unsigned*)(itc + rb + m);
          bsu[h*4+v] = *(const unsigned*)(its + rb + m);
        }
    }
    v16h bc = __builtin_bit_cast(v16h, bcu);
    v16h bs = __builtin_bit_cast(v16h, bsu);
    acc = wmma_f16(ar, bc, acc);
    acc = wmma_f16(ai, bs, acc);                   // its already holds -w*sin
  }
  if (colt < n){
    float* p = out + ((long)b*n + colt)*CKD + it*16 + (lane >> 4)*8;
    *(v8f*)p = acc;                                // (M=i contiguous) 32B-aligned
  }
}

// ---------- 6. coarsest-level linear: y = x @ T0_w.T + T0_b ----------
__global__ void t0_kernel(const float* __restrict__ x, const float* __restrict__ w,
                          const float* __restrict__ bias, float* __restrict__ y){
  int idx = blockIdx.x*blockDim.x + threadIdx.x;   // b*64 + c*8 + k
  if (idx >= NB*CKD) return;
  int k = idx & 7, bc = idx >> 3;
  const float* xi = x + bc*8;
  float acc = bias[k];
#pragma unroll
  for (int j = 0; j < 8; ++j) acc += xi[j]*w[k*8 + j];
  y[idx] = acc;
}

// ---------- 7. reconstruction: x+=Us; xc=[x,Ud]; interleave(xc@rc_e, xc@rc_o) ----------
__global__ void recon_kernel(const float* __restrict__ xin, const float* __restrict__ ud,
                             const float* __restrict__ us,
                             const float* __restrict__ rce, const float* __restrict__ rco,
                             float* __restrict__ xout, int n){
  __shared__ float re[128], ro[128];
  for (int i = threadIdx.x; i < 128; i += blockDim.x){ re[i]=rce[i]; ro[i]=rco[i]; }
  __syncthreads();
  long total = (long)NB * n * 8;
  for (long idx = (long)blockIdx.x*blockDim.x + threadIdx.x; idx < total;
       idx += (long)gridDim.x*blockDim.x){
    int  c  = (int)(idx & 7);
    long bt = idx >> 3;
    int  t  = (int)(bt % n);
    int  b  = (int)(bt / n);
    long off = ((long)b*n + t)*CKD + c*8;
    float a[16];
#pragma unroll
    for (int j = 0; j < 8; ++j){ a[j] = xin[off+j] + us[off+j]; a[8+j] = ud[off+j]; }
    float ev[8] = {0,0,0,0,0,0,0,0}, ov[8] = {0,0,0,0,0,0,0,0};
#pragma unroll
    for (int j = 0; j < 16; ++j)
#pragma unroll
      for (int k = 0; k < 8; ++k){ ev[k] += a[j]*re[j*8+k]; ov[k] += a[j]*ro[j*8+k]; }
    long ob = ((long)b*(2*n) + 2*t)*CKD + c*8;
#pragma unroll
    for (int k = 0; k < 8; ++k){ xout[ob+k] = ev[k]; xout[ob+CKD+k] = ov[k]; }
  }
}

// ================= host orchestration =================
extern "C" void kernel_launch(void* const* d_in, const int* in_sizes, int n_in,
                              void* d_out, int out_size, void* d_ws, size_t ws_size,
                              hipStream_t stream){
  (void)in_sizes; (void)n_in; (void)out_size; (void)ws_size;
  const float* x_in = (const float*)d_in[0];
  const float* wAre = (const float*)d_in[1];
  const float* wAim = (const float*)d_in[2];
  const float* wBre = (const float*)d_in[3];
  const float* wBim = (const float*)d_in[4];
  const float* wCre = (const float*)d_in[5];
  const float* wCim = (const float*)d_in[6];
  const float* T0w  = (const float*)d_in[7];
  const float* T0b  = (const float*)d_in[8];
  const float* ec_s = (const float*)d_in[9];
  const float* ec_d = (const float*)d_in[10];
  const float* rc_e = (const float*)d_in[11];
  const float* rc_o = (const float*)d_in[12];
  float* out = (float*)d_out;

  // ---- workspace carve-up ----
  char* ws = (char*)d_ws;
  size_t cur = 0;
  auto alloc = [&](size_t bytes)->char*{
    char* p = ws + cur; cur = (cur + bytes + 255) & ~(size_t)255; return p; };
  const size_t xhalf = (size_t)NB * (NN/2) * CKD;          // 8.39M floats
  float* sA   = (float*)alloc(xhalf*4);
  float* sB   = (float*)alloc(xhalf*4);
  float* dbuf = (float*)alloc(xhalf*4);
  const size_t spec  = (size_t)NB * 64 * 64;               // one spectrum
  const size_t specz = spec * KSPLIT;                      // with K-split partials
  float* DFre  = (float*)alloc(specz*4); float* DFim  = (float*)alloc(specz*4);
  float* SFre  = (float*)alloc(specz*4); float* SFim  = (float*)alloc(specz*4);
  float* UdFre = (float*)alloc(spec*4);  float* UdFim = (float*)alloc(spec*4);
  float* UsFre = (float*)alloc(spec*4);  float* UsFim = (float*)alloc(spec*4);
  const size_t tabelems = (size_t)64 * (NN - 1);           // sum of 64*n over levels
  unsigned short* ftc = (unsigned short*)alloc(tabelems*2 + 256);
  unsigned short* fts = (unsigned short*)alloc(tabelems*2 + 256);
  unsigned short* itc = (unsigned short*)alloc(tabelems*2 + 256);
  unsigned short* its = (unsigned short*)alloc(tabelems*2 + 256);

  // ---- output slot offsets (x, Ud[0..12], Us[0..12]) ----
  size_t ud_off[NLV], us_off[NLV];
  size_t co = (size_t)NB * NN * CKD;
  for (int l = 0; l < NLV; ++l){ ud_off[l] = co; co += (size_t)NB*(NN >> (l+1))*CKD; }
  for (int l = 0; l < NLV; ++l){ us_off[l] = co; co += (size_t)NB*(NN >> (l+1))*CKD; }

  // ---- 1. build DFT basis tables (deterministic, rebuilt each call) ----
  for (int l = 0; l < NLV; ++l){
    int n = NN >> (l+1);
    int lmodes = (64 < n/2 + 1) ? 64 : (n/2 + 1);
    size_t toff = (size_t)64 * (NN - (NN >> l));
    int total = 64*n;
    int blocks = (total + 255)/256; if (blocks > 2048) blocks = 2048;
    tables_kernel<<<blocks, 256, 0, stream>>>(ftc+toff, fts+toff, itc+toff, its+toff,
                                              n, lmodes);
  }

  // ---- 2. decomposition ladder ----
  const float* xcur = x_in;
  for (int l = 0; l < NLV; ++l){
    int n = NN >> (l+1);
    float* sbuf = (l & 1) ? sB : sA;
    long total = (long)NB*n*8;
    int blocks = (int)((total + 255)/256); if (blocks < 1) blocks = 1;
    decompose_kernel<<<blocks, 256, 0, stream>>>(xcur, ec_d, ec_s, dbuf, sbuf, n);

    size_t toff = (size_t)64 * (NN - (NN >> l));
    fwd_dft_kernel<<<dim3(NB, 2, KSPLIT), 256, 0, stream>>>(
        dbuf, sbuf, ftc+toff, fts+toff, DFre, DFim, SFre, SFim, n);
    mix_kernel<<<dim3(NB, 64), 64, 0, stream>>>(DFre, DFim, SFre, SFim,
                                                wAre, wAim, wBre, wBim, wCre, wCim,
                                                UdFre, UdFim, UsFre, UsFim);
    inv_dft_kernel<<<dim3(NB, 2, (n + 31)/32), 256, 0, stream>>>(
        UdFre, UdFim, UsFre, UsFim, out + ud_off[l], out + us_off[l],
        itc+toff, its+toff, n);
    xcur = sbuf;
  }

  // ---- 3. coarsest linear (s12 lives in sA since level 12 is even) ----
  t0_kernel<<<(NB*CKD + 255)/256, 256, 0, stream>>>(xcur, T0w, T0b, dbuf);

  // ---- 4. reconstruction ladder ----
  float* rcur = dbuf;
  for (int i = NLV - 1; i >= 0; --i){
    int n = NN >> (i+1);
    float* rout = (i == 0) ? out : ((rcur == sA) ? sB : sA);
    long total = (long)NB*n*8;
    int blocks = (int)((total + 255)/256); if (blocks < 1) blocks = 1;
    recon_kernel<<<blocks, 256, 0, stream>>>(rcur, out + ud_off[i], out + us_off[i],
                                             rc_e, rc_o, rout, n);
    rcur = rout;
  }
}
